// DifferentiableTree_31147102831184
// MI455X (gfx1250) — matrix operand order repaired and checked by reading, not compiled, not run on hardware
//
#include <hip/hip_runtime.h>
#include <hip/hip_bf16.h>
#include <math.h>

typedef __attribute__((ext_vector_type(16))) __bf16 v16bf;
typedef __attribute__((ext_vector_type(8)))  __bf16 v8bf;
typedef __attribute__((ext_vector_type(8)))  float  v8f;

#define N_SAMPLES 65536
#define NFEAT     64
#define NCOL      128      // 2*NFEAT
#define DEPTH     10
#define NINT      1023
#define NLEAF     1024
#define NCLS      128
#define TM        16       // samples per block (one WMMA M-tile)
#define XS_STRIDE 132      // padded X tile row stride (floats)
#define W0_STRIDE 520      // padded even-level w buffer stride (floats)
#define W1_STRIDE 264      // padded odd-level w buffer stride (floats)
#define A_STRIDE  1032     // padded A-tile row stride (bf16) -> conflict-free ds_load_b128
#define P_STRIDE  132      // padded pred tile stride (floats)
#define SQRT2_F   1.41421356237309504880f

// ---------------------------------------------------------------------------
// Prep: transpose leaf_values (1024x128 f32, row-major) into Bt[class][leaf]
// as bf16 hi/lo split (128x1024 each). Coalesced bf16 writes; f32 reads are
// L2-resident (leaf_values is only 512 KB).
// ---------------------------------------------------------------------------
__global__ __launch_bounds__(256) void bt_prep_kernel(const float* __restrict__ leaf,
                                                      __bf16* __restrict__ btHi,
                                                      __bf16* __restrict__ btLo) {
    int idx = blockIdx.x * 256 + threadIdx.x;   // 0 .. 128*1024-1
    int c = idx >> 10;                          // class
    int l = idx & (NLEAF - 1);                  // leaf
    float v = leaf[l * NCLS + c];
    __bf16 h = (__bf16)v;
    btHi[idx] = h;
    btLo[idx] = (__bf16)(v - (float)h);
}

// ---------------------------------------------------------------------------
// Fused tree kernel: gates -> leaf weights (LDS) -> bf16x3 WMMA GEMM -> norm.
// One block = 16 samples, 256 threads = 8 wave32 (one N-tile of 16 classes
// per wave).
// ---------------------------------------------------------------------------
__global__ __launch_bounds__(256) void tree_kernel(const float* __restrict__ X,
                                                   const int*   __restrict__ feat,
                                                   const float* __restrict__ thr,
                                                   const __bf16* __restrict__ btHi,
                                                   const __bf16* __restrict__ btLo,
                                                   float* __restrict__ out) {
    __shared__ float  Xs[TM * XS_STRIDE];                    //  8.4 KB
    __shared__ int    featS[NINT];                           //  4.1 KB
    __shared__ float  thrS[NINT];                            //  4.1 KB
    __shared__ float  wbuf0[TM * W0_STRIDE];                 // 33.3 KB (even levels, max 512)
    __shared__ float  wbuf1[TM * W1_STRIDE];                 // 16.9 KB (odd levels,  max 256)
    __shared__ alignas(16) __bf16 Ahi[TM * A_STRIDE];        // 33.0 KB
    __shared__ alignas(16) __bf16 Alo[TM * A_STRIDE];        // 33.0 KB

    const int tid = threadIdx.x;
    const int n0  = blockIdx.x * TM;

    // ---- stage X tile + tree parameters in LDS (fully coalesced) ----
    #pragma unroll
    for (int i = 0; i < (TM * NCOL) / 256; ++i) {
        int e = tid + 256 * i;
        int r = e >> 7, c = e & (NCOL - 1);
        Xs[r * XS_STRIDE + c] = X[(n0 + r) * NCOL + c];
    }
    for (int j = tid; j < NINT; j += 256) { featS[j] = feat[j]; thrS[j] = thr[j]; }
    __syncthreads();

    // ---- hierarchical expansion: level l reads 2^l weights, writes 2^(l+1).
    //      Gates computed on the fly: exactly TM*1023 erf evals per block. ----
    for (int l = 0; l < DEPTH; ++l) {
        float*       wout = (l & 1) ? wbuf1 : wbuf0;
        const float* win  = (l & 1) ? wbuf0 : wbuf1;
        const int    so   = (l & 1) ? W1_STRIDE : W0_STRIDE;   // out stride
        const int    si   = (l & 1) ? W0_STRIDE : W1_STRIDE;   // in stride
        const int    mask = (1 << l) - 1;
        const int    T    = TM << l;
        for (int t = tid; t < T; t += 256) {
            int   i    = t & mask;
            int   s    = t >> l;
            int   node = mask + i;                // heap index at this level
            int   f    = featS[node];
            float mean = Xs[s * XS_STRIDE + f];
            float sd   = Xs[s * XS_STRIDE + NFEAT + f];
            float z    = (thrS[node] - mean) / (sd * SQRT2_F);
            float g    = 0.5f * (1.0f + erff(z));
            float wp   = (l == 0) ? 1.0f : win[s * si + i];
            float c0   = wp * g;
            float c1   = wp * (1.0f - g);
            if (l < DEPTH - 1) {
                wout[s * so + 2 * i]     = c0;
                wout[s * so + 2 * i + 1] = c1;
            } else {
                // final leaf weights -> bf16 hi/lo A-tile (row-major per sample)
                __bf16 h0 = (__bf16)c0;
                __bf16 h1 = (__bf16)c1;
                Ahi[s * A_STRIDE + 2 * i]     = h0;
                Ahi[s * A_STRIDE + 2 * i + 1] = h1;
                Alo[s * A_STRIDE + 2 * i]     = (__bf16)(c0 - (float)h0);
                Alo[s * A_STRIDE + 2 * i + 1] = (__bf16)(c1 - (float)h1);
            }
        }
        __syncthreads();
    }

    // ---- GEMM: pred[16x128] = w[16x1024] @ leaf_values[1024x128]
    //      wave wv owns classes [wv*16, wv*16+16); bf16x3 for ~f32 accuracy ----
    const int lane  = tid & 31;
    const int wv    = tid >> 5;                    // 0..7
    const int arow  = lane & 15;                   // A: M = lane%16
    const int akoff = (lane < 16) ? 0 : 16;        // A: K run of 16 at akoff
    const int cls   = wv * 16 + (lane & 15);       // B: N = class
    const int bkoff = (lane < 16) ? 0 : 16;        // B: K run of 16

    const __bf16* aHiB = &Ahi[arow * A_STRIDE + akoff];
    const __bf16* aLoB = &Alo[arow * A_STRIDE + akoff];
    const __bf16* bHiB = btHi + cls * NLEAF + bkoff;
    const __bf16* bLoB = btLo + cls * NLEAF + bkoff;

    v8f acc = {};
    #pragma unroll 4
    for (int kc = 0; kc < NLEAF; kc += 32) {
        union { v16bf v; v8bf h[2]; } ah, al;
        ah.h[0] = *(const v8bf*)(aHiB + kc);        // K = kc+akoff    .. +7
        ah.h[1] = *(const v8bf*)(aHiB + kc + 8);    // K = kc+akoff+8  .. +15
        al.h[0] = *(const v8bf*)(aLoB + kc);
        al.h[1] = *(const v8bf*)(aLoB + kc + 8);
        v16bf bh = *(const v16bf*)(bHiB + kc);      // 16 contiguous K for this class
        v16bf bl = *(const v16bf*)(bLoB + kc);
        acc = __builtin_amdgcn_wmma_f32_16x16x32_bf16(false, ah.v, false, bh,
                                                      (short)0, acc, false, false);
        acc = __builtin_amdgcn_wmma_f32_16x16x32_bf16(false, ah.v, false, bl,
                                                      (short)0, acc, false, false);
        acc = __builtin_amdgcn_wmma_f32_16x16x32_bf16(false, al.v, false, bh,
                                                      (short)0, acc, false, false);
    }

    // ---- scatter C tile to LDS (reuse dead wbuf0/wbuf1), row-normalize, store ----
    float* predS = wbuf0;                 // 16 x P_STRIDE floats (<= wbuf0 size)
    float* partS = wbuf1;                 // 16 x 16 partials
    float* invS  = wbuf1 + 256;           // 16 reciprocals
    {
        int rbase = (lane < 16) ? 0 : 8;  // C layout: VGPR r -> M = r (+8 for hi lanes)
        int col   = wv * 16 + (lane & 15);
        #pragma unroll
        for (int r = 0; r < 8; ++r)
            predS[(rbase + r) * P_STRIDE + col] = acc[r];
    }
    __syncthreads();
    {
        int r = tid >> 4, j = tid & 15;   // 16x16 partial row sums
        float s = 0.0f;
        #pragma unroll
        for (int c = 0; c < 8; ++c) s += predS[r * P_STRIDE + j * 8 + c];
        partS[r * 16 + j] = s;
    }
    __syncthreads();
    if (tid < TM) {
        float s = 0.0f;
        #pragma unroll
        for (int j = 0; j < 16; ++j) s += partS[tid * 16 + j];
        invS[tid] = 1.0f / s;
    }
    __syncthreads();
    #pragma unroll
    for (int i = 0; i < (TM * NCLS) / 256; ++i) {
        int e = tid + 256 * i;
        int r = e >> 7, c = e & (NCLS - 1);
        out[(n0 + r) * NCLS + c] = predS[r * P_STRIDE + c] * invS[r];
    }
}

// ---------------------------------------------------------------------------
extern "C" void kernel_launch(void* const* d_in, const int* in_sizes, int n_in,
                              void* d_out, int out_size, void* d_ws, size_t ws_size,
                              hipStream_t stream) {
    const float* X    = (const float*)d_in[0];   // 65536 x 128 f32
    const int*   feat = (const int*)  d_in[1];   // 1023 i32
    const float* thr  = (const float*)d_in[2];   // 1023 f32
    const float* leaf = (const float*)d_in[3];   // 1024 x 128 f32
    float*       out  = (float*)d_out;           // 65536 x 128 f32

    __bf16* btHi = (__bf16*)d_ws;                // 128 x 1024 bf16 (256 KB)
    __bf16* btLo = btHi + NCLS * NLEAF;          // 128 x 1024 bf16 (256 KB)

    bt_prep_kernel<<<(NCLS * NLEAF) / 256, 256, 0, stream>>>(leaf, btHi, btLo);
    tree_kernel<<<N_SAMPLES / TM, 256, 0, stream>>>(X, feat, thr, btHi, btLo, out);
}
